// SparseAttention_83940840833230
// MI455X (gfx1250) — matrix operand order repaired and checked by reading, not compiled, not run on hardware
//
#include <hip/hip_runtime.h>

typedef __attribute__((ext_vector_type(16))) __bf16 v16bf;
typedef __attribute__((ext_vector_type(8)))  float v8f;

#define S_LEN 8192
#define NB 128
#define BS 64
#define DIM 64
#define NH 16
#define STR 72   // padded LDS row stride in bf16 elements (144 bytes, 16B-aligned)

union Frag16 { v16bf v; uint4 q[2]; unsigned short s[16]; __bf16 b[16]; };
union Acc8   { v8f v; float f[8]; };

__device__ __forceinline__ unsigned short f2bf(float f) {
  __bf16 h = (__bf16)f;                       // native RNE convert on gfx1250
  return __builtin_bit_cast(unsigned short, h);
}
__device__ __forceinline__ unsigned pk2(float a, float b) {
  return (unsigned)f2bf(a) | ((unsigned)f2bf(b) << 16);   // v_cvt_pk_bf16_f32 path
}

__global__ __launch_bounds__(128) void sparse_attn_wmma(
    const float* __restrict__ qg, const float* __restrict__ kg,
    const float* __restrict__ vg, float* __restrict__ outg)
{
  __shared__ __align__(16) unsigned short lK[BS * STR];        // K block, row-major (key, dim)
  __shared__ __align__(16) unsigned short lV[BS * STR];        // V block, transposed (dim, key)
  __shared__ __align__(16) unsigned short lP[4 * 16 * STR];    // per-wave P tiles (16 x 64)

  const int i    = blockIdx.x;           // query block
  const int h    = blockIdx.y;           // head
  const int tid  = threadIdx.x;
  const int wave = tid >> 5;
  const int lane = tid & 31;
  const int half = lane >> 4;
  const int ln   = lane & 15;
  const size_t hb = (size_t)h * S_LEN * DIM;

  // Scores computed directly in log2 domain: scale = (1/sqrt(D)) * log2(e)
  const float QSCALE = 0.125f * 1.4426950408889634f;

  // ---------------- Q fragments (A-layout), pre-scaled ----------------
  Frag16 qa[2];
  {
    const float* qp = qg + hb + (size_t)(i * BS + wave * 16 + ln) * DIM;
    #pragma unroll
    for (int ks = 0; ks < 2; ++ks) {
      const float4* c0 = (const float4*)(qp + ks * 32 + half * 8);
      const float4* c1 = (const float4*)(qp + ks * 32 + 16 + half * 8);
      float4 a = c0[0], b = c0[1], c = c1[0], d = c1[1];
      float t0[8] = {a.x, a.y, a.z, a.w, b.x, b.y, b.z, b.w};
      float t1[8] = {c.x, c.y, c.z, c.w, d.x, d.y, d.z, d.w};
      #pragma unroll
      for (int e = 0; e < 8; ++e) {
        qa[ks].b[e]     = (__bf16)(t0[e] * QSCALE);
        qa[ks].b[8 + e] = (__bf16)(t1[e] * QSCALE);
      }
    }
  }

  // ---------------- flash-softmax state (log2 domain) ----------------
  Acc8 o[4];
  float rm[8], rl[8];
  #pragma unroll
  for (int g = 0; g < 8; ++g) { rm[g] = -1e30f; rl[g] = 0.0f; }
  #pragma unroll
  for (int nt = 0; nt < 4; ++nt)
    #pragma unroll
    for (int g = 0; g < 8; ++g) o[nt].f[g] = 0.0f;

  const int ncols = (i < 8) ? (i + 1) : 9;   // sink block 0 + local window
  for (int c = 0; c < ncols; ++c) {
    const int j = (i < 8) ? c : (c ? (i - 8 + c) : 0);

    __syncthreads();   // previous iteration's lK/lV reads complete
    // ---------------- cooperative stage: K (row-major) and V^T into LDS as bf16 ----
    {
      const int r  = tid >> 1;           // key row 0..63
      const int ch = tid & 1;            // dim half: 32 floats
      const float* kp = kg + hb + (size_t)(j * BS + r) * DIM + ch * 32;
      const float* vp = vg + hb + (size_t)(j * BS + r) * DIM + ch * 32;
      float4 kf[8], vf[8];
      #pragma unroll
      for (int t = 0; t < 8; ++t) { kf[t] = ((const float4*)kp)[t]; vf[t] = ((const float4*)vp)[t]; }
      uint4* kd = (uint4*)(lK + r * STR + ch * 32);
      #pragma unroll
      for (int t = 0; t < 4; ++t) {
        uint4 w;
        w.x = pk2(kf[2*t].x,   kf[2*t].y);   w.y = pk2(kf[2*t].z,   kf[2*t].w);
        w.z = pk2(kf[2*t+1].x, kf[2*t+1].y); w.w = pk2(kf[2*t+1].z, kf[2*t+1].w);
        kd[t] = w;
      }
      #pragma unroll
      for (int t = 0; t < 8; ++t) {
        float fv[4] = {vf[t].x, vf[t].y, vf[t].z, vf[t].w};
        #pragma unroll
        for (int e = 0; e < 4; ++e) {
          int d0 = ch * 32 + t * 4 + e;
          lV[d0 * STR + r] = f2bf(fv[e]);
        }
      }
    }
    __syncthreads();

    // ---------------- S = Q K^T (8 WMMAs), S already in log2 units -------------
    Acc8 s[4];
    #pragma unroll
    for (int nt = 0; nt < 4; ++nt) {
      v8f acc;
      #pragma unroll
      for (int g = 0; g < 8; ++g) acc[g] = 0.0f;
      #pragma unroll
      for (int ks = 0; ks < 2; ++ks) {
        Frag16 kb;   // B-layout: lane = key column, K chunks at 8*half and 16+8*half
        const uint4* src = (const uint4*)(lK + (nt * 16 + ln) * STR + ks * 32 + half * 8);
        kb.q[0] = src[0]; kb.q[1] = src[2];
        acc = __builtin_amdgcn_wmma_f32_16x16x32_bf16(false, qa[ks].v, false, kb.v,
                                                      (short)0, acc, false, false);
      }
      s[nt].v = acc;
    }

    // ---------------- causal mask on diagonal block ----------------
    if (j == i) {
      #pragma unroll
      for (int nt = 0; nt < 4; ++nt)
        #pragma unroll
        for (int g = 0; g < 8; ++g)
          if (nt * 16 + ln > wave * 16 + g + 8 * half) s[nt].f[g] = -1e30f;
    }

    // ---------------- online softmax, exp2 domain (row = g + 8*half) -----------
    #pragma unroll
    for (int g = 0; g < 8; ++g) {
      float tm = fmaxf(fmaxf(s[0].f[g], s[1].f[g]), fmaxf(s[2].f[g], s[3].f[g]));
      tm = fmaxf(tm, __shfl_xor(tm, 1));
      tm = fmaxf(tm, __shfl_xor(tm, 2));
      tm = fmaxf(tm, __shfl_xor(tm, 4));
      tm = fmaxf(tm, __shfl_xor(tm, 8));
      float nm = fmaxf(rm[g], tm);
      float sc = __builtin_amdgcn_exp2f(rm[g] - nm);   // bare v_exp_f32
      rm[g] = nm;
      float ts = 0.0f;
      #pragma unroll
      for (int nt = 0; nt < 4; ++nt) {
        float p = __builtin_amdgcn_exp2f(s[nt].f[g] - nm);
        s[nt].f[g] = p;
        ts += p;
      }
      ts += __shfl_xor(ts, 1);
      ts += __shfl_xor(ts, 2);
      ts += __shfl_xor(ts, 4);
      ts += __shfl_xor(ts, 8);
      rl[g] = rl[g] * sc + ts;
      #pragma unroll
      for (int nt = 0; nt < 4; ++nt) o[nt].f[g] *= sc;
    }

    // ---------------- P -> per-wave LDS (bf16), then O += P V (8 WMMAs) --------
    unsigned short* pb = lP + wave * 16 * STR;
    #pragma unroll
    for (int nt = 0; nt < 4; ++nt)
      #pragma unroll
      for (int g = 0; g < 8; ++g)
        pb[(g + 8 * half) * STR + nt * 16 + ln] = f2bf(s[nt].f[g]);
    // LDS ops are in-order within a wave (DScnt), so reads below see the writes.
    #pragma unroll
    for (int ks = 0; ks < 2; ++ks) {
      Frag16 pa;   // A-layout: lane = query row ln, K chunks at 8*half and 16+8*half
      pa.q[0] = *(const uint4*)(pb + ln * STR + ks * 32 + half * 8);
      pa.q[1] = *(const uint4*)(pb + ln * STR + ks * 32 + 16 + half * 8);
      #pragma unroll
      for (int nt = 0; nt < 4; ++nt) {
        Frag16 vb;   // B-layout from V^T: lane = dim column, K chunks at 8*half and 16+8*half
        const uint4* src = (const uint4*)(lV + (nt * 16 + ln) * STR + ks * 32 + half * 8);
        vb.q[0] = src[0]; vb.q[1] = src[2];
        o[nt].v = __builtin_amdgcn_wmma_f32_16x16x32_bf16(false, pa.v, false, vb.v,
                                                          (short)0, o[nt].v, false, false);
      }
    }
  }

  // ---------------- epilogue: O * (1/l) via v_rcp_f32, coalesced stores ----------
  float inv[8];
  #pragma unroll
  for (int g = 0; g < 8; ++g) inv[g] = __builtin_amdgcn_rcpf(rl[g]);
  float* op = outg + hb + (size_t)(i * BS + wave * 16) * DIM;
  #pragma unroll
  for (int nt = 0; nt < 4; ++nt)
    #pragma unroll
    for (int g = 0; g < 8; ++g)
      op[(size_t)(g + 8 * half) * DIM + nt * 16 + ln] = o[nt].f[g] * inv[g];
}

extern "C" void kernel_launch(void* const* d_in, const int* in_sizes, int n_in,
                              void* d_out, int out_size, void* d_ws, size_t ws_size,
                              hipStream_t stream) {
  (void)in_sizes; (void)n_in; (void)out_size; (void)d_ws; (void)ws_size;
  const float* q = (const float*)d_in[0];
  const float* k = (const float*)d_in[1];
  const float* v = (const float*)d_in[2];
  float* out = (float*)d_out;
  dim3 grid(NB, NH, 1);
  sparse_attn_wmma<<<grid, 128, 0, stream>>>(q, k, v, out);
}
